// Transformer_encoder_89292370084470
// MI455X (gfx1250) — matrix operand order repaired and checked by reading, not compiled, not run on hardware
//
#include <hip/hip_runtime.h>

// ---------------------------------------------------------------------------
// Types for CDNA5 WMMA (wave32): v_wmma_f32_16x16x32_bf16
// ---------------------------------------------------------------------------
typedef __attribute__((ext_vector_type(16))) __bf16          bf16x16;
typedef __attribute__((ext_vector_type(16))) unsigned short  u16x16;
typedef __attribute__((ext_vector_type(8)))  float           f32x8;
typedef __attribute__((ext_vector_type(4)))  unsigned int    uint32x4;
typedef __attribute__((ext_vector_type(4)))  int             int32x4;
typedef __attribute__((ext_vector_type(8)))  int             int32x8;

__device__ __forceinline__ unsigned short f2bf(float f) {
  unsigned int x = __float_as_uint(f);
  x += 0x7fffu + ((x >> 16) & 1u);          // round-to-nearest-even
  return (unsigned short)(x >> 16);
}

__device__ __forceinline__ f32x8 wmma_bf16(bf16x16 a, bf16x16 b, f32x8 c) {
  return __builtin_amdgcn_wmma_f32_16x16x32_bf16(
      /*neg_a=*/false, a, /*neg_b=*/false, b,
      /*c_mod=*/(short)0, c, /*reuse_a=*/false, /*reuse_b=*/false);
}

// A-matrix 16x32 bf16 fragment from row-major f32 [rows, ld], window (row0, k0)
__device__ __forceinline__ bf16x16 load_a_f32(const float* base, int ld,
                                              int row0, int k0, int lane) {
  int m = lane & 15, hi = (lane >> 4) & 1;
  const float* p = base + (size_t)(row0 + m) * ld + k0;
  u16x16 u;
#pragma unroll
  for (int r = 0; r < 8; ++r) {
    int k = 2 * r + 16 * hi;
    u[2 * r]     = f2bf(p[k]);
    u[2 * r + 1] = f2bf(p[k + 1]);
  }
  return __builtin_bit_cast(bf16x16, u);
}

// A-matrix 16x32 bf16 fragment from LDS-resident bf16 (ushort) rows [16, ld]
__device__ __forceinline__ bf16x16 load_a_lds(const unsigned short* base, int ld,
                                              int k0, int lane) {
  int m = lane & 15, hi = (lane >> 4) & 1;
  u16x16 u;
#pragma unroll
  for (int r = 0; r < 8; ++r) {
    int k = 2 * r + 16 * hi;
    u[2 * r]     = base[m * ld + k0 + k];
    u[2 * r + 1] = base[m * ld + k0 + k + 1];
  }
  return __builtin_bit_cast(bf16x16, u);
}

// B-matrix 32x16 bf16 fragment from row-major f32 W[K, ld]; B[k][n] = W[k0+k][col0+n]
__device__ __forceinline__ bf16x16 load_b_f32(const float* base, int ld,
                                              int k0, int col0, int lane) {
  int n = lane & 15, hi = (lane >> 4) & 1;
  u16x16 u;
#pragma unroll
  for (int r = 0; r < 8; ++r) {
    int k = k0 + 16 * hi + 2 * r;
    u[2 * r]     = f2bf(base[(size_t)k * ld + col0 + n]);
    u[2 * r + 1] = f2bf(base[(size_t)(k + 1) * ld + col0 + n]);
  }
  return __builtin_bit_cast(bf16x16, u);
}

// B-matrix 32x16 bf16 fragment, transposed source: B[k][n] = M[(j0+n)*ld + k0 + k]
__device__ __forceinline__ bf16x16 load_bT_f32(const float* base, int ld,
                                               int k0, int j0, int lane) {
  int n = lane & 15, hi = (lane >> 4) & 1;
  const float* p = base + (size_t)(j0 + n) * ld + k0 + 16 * hi;
  u16x16 u;
#pragma unroll
  for (int r = 0; r < 8; ++r) {
    u[2 * r]     = f2bf(p[2 * r]);
    u[2 * r + 1] = f2bf(p[2 * r + 1]);
  }
  return __builtin_bit_cast(bf16x16, u);
}

// ---------------------------------------------------------------------------
// Tensor Data Mover: 2-D tile load Global -> LDS (D# built per ISA cdna5 ch.8)
// ---------------------------------------------------------------------------
__device__ __forceinline__ void tdm_load_2d(const void* gaddr, unsigned lds_off,
                                            unsigned tile_w, unsigned tile_h,
                                            unsigned row_stride_elems) {
  unsigned long long ga = (unsigned long long)gaddr;
  uint32x4 g0;
  g0[0] = 1u;                                         // count=1, user-mode, no gather
  g0[1] = lds_off;                                    // lds_addr (bytes)
  g0[2] = (unsigned)(ga & 0xffffffffu);               // global_addr[31:0]
  g0[3] = (unsigned)((ga >> 32) & 0x01ffffffu)        // global_addr[56:32]
          | (2u << 30);                               // type = 2 (image)
  int32x8 g1;
  g1[0] = (int)(2u << 16);                            // wg_mask=0, data_size=2 (4B)
  g1[1] = (int)((tile_w & 0xffffu) << 16);            // tensor_dim0[15:0] = tile_w
  g1[2] = (int)((tile_h & 0xffffu) << 16);            // tensor_dim0 hi=0 | tensor_dim1[15:0]
  g1[3] = (int)((tile_w & 0xffffu) << 16);            // tensor_dim1 hi=0 | tile_dim0
  g1[4] = (int)(tile_h & 0xffffu);                    // tile_dim1 | tile_dim2=0
  g1[5] = (int)row_stride_elems;                      // tensor_dim0_stride[31:0]
  g1[6] = 0;                                          // stride hi | dim1_stride lo
  g1[7] = 0;
  int32x4 gz = {0, 0, 0, 0};                          // 2-D tensor: groups 2/3 unused
#if defined(__clang_major__) && (__clang_major__ >= 23)
  int32x8 gz8 = {0, 0, 0, 0, 0, 0, 0, 0};
  __builtin_amdgcn_tensor_load_to_lds(g0, g1, gz, gz, gz8, 0);
#else
  __builtin_amdgcn_tensor_load_to_lds(g0, g1, gz, gz, 0);
#endif
}

// ---------------------------------------------------------------------------
// Utility kernels
// ---------------------------------------------------------------------------
__global__ void zero_k(float* p, int n) {
  int i = blockIdx.x * blockDim.x + threadIdx.x;
  if (i < n) p[i] = 0.f;
}

__global__ void add3_k(const float* a, const float* b, const float* c, float* o, int n) {
  int i = blockIdx.x * blockDim.x + threadIdx.x;
  if (i < n) o[i] = a[i] + b[i] + c[i];
}

__global__ void add2_k(const float* a, const float* b, float* o, int n) {
  int i = blockIdx.x * blockDim.x + threadIdx.x;
  if (i < n) o[i] = a[i] + b[i];
}

// ---------------------------------------------------------------------------
// PE BatchNorm stats (20 features / 4096 rows) -> mv[0..19]=mean, mv[32..51]=var
// ---------------------------------------------------------------------------
__global__ void pe_stats_k(const float* __restrict__ pe, float* __restrict__ mv) {
  __shared__ float ssum[20], ssq[20];
  int t = threadIdx.x;
  if (t < 20) { ssum[t] = 0.f; ssq[t] = 0.f; }
  __syncthreads();
  float ls[20], lq[20];
#pragma unroll
  for (int i = 0; i < 20; ++i) { ls[i] = 0.f; lq[i] = 0.f; }
  for (int r = t; r < 4096; r += 256) {
#pragma unroll
    for (int i = 0; i < 20; ++i) {
      float v = pe[r * 20 + i];
      ls[i] += v; lq[i] += v * v;
    }
  }
#pragma unroll
  for (int i = 0; i < 20; ++i) { atomicAdd(&ssum[i], ls[i]); atomicAdd(&ssq[i], lq[i]); }
  __syncthreads();
  if (t < 20) {
    float m = ssum[t] * (1.f / 4096.f);
    mv[t]      = m;
    mv[32 + t] = ssq[t] * (1.f / 4096.f) - m * m;   // biased variance
  }
}

// ---------------------------------------------------------------------------
// Node embedding: x = [pos @ W1 + b1 | BN(pe) @ W2 + b2],  x:[4096,256]
// ---------------------------------------------------------------------------
__global__ void node_embed_k(const float* __restrict__ pos, const float* __restrict__ pe,
                             const float* __restrict__ w1, const float* __restrict__ b1,
                             const float* __restrict__ w2, const float* __restrict__ b2,
                             const float* __restrict__ g,  const float* __restrict__ bb,
                             const float* __restrict__ mv, float* __restrict__ x) {
  int r = blockIdx.x, c = threadIdx.x;   // block=256
  float acc;
  if (c < 192) {
    acc = b1[c];
#pragma unroll
    for (int i = 0; i < 3; ++i) acc += pos[r * 3 + i] * w1[i * 192 + c];
  } else {
    int c2 = c - 192;
    acc = b2[c2];
#pragma unroll
    for (int i = 0; i < 20; ++i) {
      float v = (pe[r * 20 + i] - mv[i]) * rsqrtf(mv[32 + i] + 1e-5f) * g[i] + bb[i];
      acc += v * w2[i * 64 + c2];
    }
  }
  x[(size_t)r * 256 + c] = acc;
}

// ---------------------------------------------------------------------------
// Edge table: et[32,256] = edge_emb @ We + be  (only 32 distinct edge types)
// ---------------------------------------------------------------------------
__global__ void e_table_k(const float* __restrict__ eemb, const float* __restrict__ we,
                          const float* __restrict__ be, float* __restrict__ et) {
  int a = blockIdx.x, c = threadIdx.x;   // grid 32 x block 256
  float acc = be[c];
  for (int i = 0; i < 256; ++i) acc += eemb[a * 256 + i] * we[i * 256 + c];
  et[a * 256 + c] = acc;
}

// ---------------------------------------------------------------------------
// WMMA bf16 GEMM with double-buffered TDM staging:
//   out[M,N] = epi( (A[M,K] @ W[K,N] + bias)*scale [+res] [relu] )
// grid = (N/64, M/64), block = 128 (4 waves). Wave0 streams 64x32 A tiles and
// 32x64 W tiles via tensor_load_to_lds into ping-pong buffers; TDM for step
// s+1 overlaps WMMA on step s (TENSORcnt completes in order per wave, so
// s_wait_tensorcnt<=2 guarantees the current pair landed).
// ---------------------------------------------------------------------------
template <bool BIAS, bool RES, bool RELU>
__global__ void __launch_bounds__(128)
gemm_t(const float* __restrict__ A, const float* __restrict__ W,
       const float* __restrict__ bias, const float* __restrict__ residual,
       float* __restrict__ out, int M, int K, int N, float scale) {
  __shared__ float As[2][64 * 32];   // 2 x 8 KB
  __shared__ float Ws[2][32 * 64];   // 2 x 8 KB
  int lane = threadIdx.x & 31;
  int wave = threadIdx.x >> 5;
  int rowB = blockIdx.y * 64;
  int colB = blockIdx.x * 64;

  f32x8 acc[4];
#pragma unroll
  for (int t = 0; t < 4; ++t) acc[t] = f32x8{0.f, 0.f, 0.f, 0.f, 0.f, 0.f, 0.f, 0.f};

  unsigned ldsA[2] = {(unsigned)(size_t)(&As[0][0]), (unsigned)(size_t)(&As[1][0])};
  unsigned ldsW[2] = {(unsigned)(size_t)(&Ws[0][0]), (unsigned)(size_t)(&Ws[1][0])};

  int S = K >> 5;                      // k-steps of 32
  if (wave == 0) {                     // prologue: stage step 0 into buffer 0
    tdm_load_2d(A + (size_t)rowB * K, ldsA[0], 32, 64, K);
    tdm_load_2d(W + (size_t)colB,     ldsW[0], 64, 32, N);
  }
  for (int s = 0; s < S; ++s) {
    int cur = s & 1;
    if (wave == 0) {
      if (s + 1 < S) {                 // prefetch step s+1 into other buffer
        int k1 = (s + 1) * 32;
        tdm_load_2d(A + (size_t)rowB * K + k1, ldsA[cur ^ 1], 32, 64, K);
        tdm_load_2d(W + (size_t)k1 * N + colB, ldsW[cur ^ 1], 64, 32, N);
        __builtin_amdgcn_s_wait_tensorcnt((short)2);  // current pair complete
      } else {
        __builtin_amdgcn_s_wait_tensorcnt((short)0);
      }
    }
    __syncthreads();                   // current tiles visible to all waves
    bf16x16 af = load_a_f32(&As[cur][0], 32, wave * 16, 0, lane);
#pragma unroll
    for (int t = 0; t < 4; ++t) {
      bf16x16 bf = load_b_f32(&Ws[cur][0], 64, 0, t * 16, lane);
      acc[t] = wmma_bf16(af, bf, acc[t]);
    }
    __syncthreads();                   // all consumers done before overwrite
  }

  int n = lane & 15, hi = (lane >> 4) & 1;
#pragma unroll
  for (int t = 0; t < 4; ++t) {
    int col0 = colB + t * 16;
    float bn = BIAS ? bias[col0 + n] : 0.f;
#pragma unroll
    for (int r = 0; r < 8; ++r) {
      int m = r + 8 * hi;
      float v = (acc[t][r] + bn) * scale;
      size_t idx = (size_t)(rowB + wave * 16 + m) * N + col0 + n;
      if (RES) v += residual[idx];
      if (RELU) v = v > 0.f ? v : 0.f;
      out[idx] = v;
    }
  }
}

// ---------------------------------------------------------------------------
// ResGatedGraphConv aggregation: agg[dst] += relu(k[dst]+q[src]+et[attr]) * v[src]
// ---------------------------------------------------------------------------
__global__ void edge_agg_k(const float* __restrict__ kk, const float* __restrict__ qq,
                           const float* __restrict__ vv, const float* __restrict__ et,
                           const int* __restrict__ eidx, const int* __restrict__ eattr,
                           float* __restrict__ agg, int E) {
  int e = blockIdx.x * 8 + (threadIdx.x >> 5);
  if (e >= E) return;
  int lane = threadIdx.x & 31;
  int s = eidx[e];
  int d = eidx[E + e];
  int a = eattr[e];
  const float* kp = kk + (size_t)d * 256;
  const float* qp = qq + (size_t)s * 256;
  const float* vp = vv + (size_t)s * 256;
  const float* ep = et + (size_t)a * 256;
  float* op = agg + (size_t)d * 256;
#pragma unroll
  for (int j = 0; j < 8; ++j) {
    int cidx = lane + j * 32;
    float g = kp[cidx] + qp[cidx] + ep[cidx];
    g = g > 0.f ? g : 0.f;
    atomicAdd(&op[cidx], g * vp[cidx]);
  }
}

// ---------------------------------------------------------------------------
// BatchNorm over axis 0: partial sums per column (atomic), then apply
// ---------------------------------------------------------------------------
__global__ void bn_stats_k(const float* __restrict__ xin, float* __restrict__ stat,
                           int M, int C, int rpb) {
  int c = threadIdx.x;                 // block = C = 256
  int r0 = blockIdx.x * rpb;
  float s = 0.f, s2 = 0.f;
  for (int r = r0; r < r0 + rpb; ++r) {
    float v = xin[(size_t)r * C + c];
    s += v; s2 += v * v;
  }
  atomicAdd(&stat[c], s);
  atomicAdd(&stat[C + c], s2);
}

__global__ void bn_apply_k(const float* __restrict__ xin, const float* __restrict__ stat,
                           const float* __restrict__ g, const float* __restrict__ bb,
                           float* __restrict__ out, int M, int C) {
  int idx = blockIdx.x * blockDim.x + threadIdx.x;
  if (idx >= M * C) return;
  int c = idx & (C - 1);
  float invM = 1.f / (float)M;
  float mean = stat[c] * invM;
  float var = stat[C + c] * invM - mean * mean;
  out[idx] = g[c] * (xin[idx] - mean) * rsqrtf(var + 1e-5f) + bb[c];
}

// ---------------------------------------------------------------------------
// Biased MHA, one wave per (b, h, 16-row tile). D=32, N=512.
// ---------------------------------------------------------------------------
__global__ void __launch_bounds__(32)
attn_k(const float* __restrict__ qh, const float* __restrict__ kh,
       const float* __restrict__ vh, const float* __restrict__ bias,
       float* __restrict__ ao) {
  __shared__ float          sS[16 * 512];   // 32 KB raw scores
  __shared__ unsigned short sP[16 * 512];   // 16 KB bf16 unnormalized probs
  __shared__ float          srow[16];       // row sums
  int id = blockIdx.x;
  int it = id & 31; id >>= 5;               // 32 row tiles
  int h  = id & 7;  id >>= 3;               // 8 heads
  int b  = id;                              // 8 graphs
  int i0 = it * 16;
  int lane = threadIdx.x;
  int n = lane & 15, hi = (lane >> 4) & 1;

  const float* qb = qh + (size_t)b * 512 * 256;
  const float* kb = kh + (size_t)b * 512 * 256;
  const float* vb = vh + (size_t)b * 512 * 256;

  bf16x16 aq = load_a_f32(qb, 256, i0, h * 32, lane);

  for (int jt = 0; jt < 32; ++jt) {
    bf16x16 bkf = load_bT_f32(kb, 256, h * 32, jt * 16, lane);
    f32x8 c = {0.f, 0.f, 0.f, 0.f, 0.f, 0.f, 0.f, 0.f};
    c = wmma_bf16(aq, bkf, c);
#pragma unroll
    for (int r = 0; r < 8; ++r) {
      int m = r + 8 * hi;
      size_t bidx = (((size_t)b * 512 + i0 + m) * 512 + jt * 16 + n) * 8 + h;
      sS[m * 512 + jt * 16 + n] = c[r] + bias[bidx];
    }
  }
  __syncthreads();

  {
    int m = lane & 15, half = (lane >> 4) & 1;
    float mx = -3.0e38f;
    for (int j = half * 256; j < half * 256 + 256; ++j)
      mx = fmaxf(mx, sS[m * 512 + j]);
    mx = fmaxf(mx, __shfl_xor(mx, 16, 32));
    float sum = 0.f;
    for (int j = half * 256; j < half * 256 + 256; ++j) {
      float p = __expf(sS[m * 512 + j] - mx);
      sum += p;
      sP[m * 512 + j] = f2bf(p);
    }
    sum += __shfl_xor(sum, 16, 32);
    if (half == 0) srow[m] = sum;
  }
  __syncthreads();

  f32x8 c0 = {0.f, 0.f, 0.f, 0.f, 0.f, 0.f, 0.f, 0.f};
  f32x8 c1 = {0.f, 0.f, 0.f, 0.f, 0.f, 0.f, 0.f, 0.f};
  for (int k0 = 0; k0 < 512; k0 += 32) {
    bf16x16 ap = load_a_lds(sP, 512, k0, lane);
    bf16x16 b0 = load_b_f32(vb, 256, k0, h * 32, lane);
    bf16x16 b1 = load_b_f32(vb, 256, k0, h * 32 + 16, lane);
    c0 = wmma_bf16(ap, b0, c0);
    c1 = wmma_bf16(ap, b1, c1);
  }
#pragma unroll
  for (int r = 0; r < 8; ++r) {
    int m = r + 8 * hi;
    float inv = 1.f / srow[m];
    size_t orow = ((size_t)b * 512 + i0 + m) * 256 + h * 32;
    ao[orow + n]      = c0[r] * inv;
    ao[orow + 16 + n] = c1[r] * inv;
  }
}

// ---------------------------------------------------------------------------
// global_mean_pool + head MLP
// ---------------------------------------------------------------------------
__global__ void pool_k(const float* __restrict__ node, float* __restrict__ pooled) {
  int b = blockIdx.x, c = threadIdx.x;
  float s = 0.f;
  for (int r = 0; r < 512; ++r) s += node[((size_t)b * 512 + r) * 256 + c];
  pooled[b * 256 + c] = s * (1.f / 512.f);
}

__global__ void head_k(const float* __restrict__ pooled,
                       const float* __restrict__ w1, const float* __restrict__ b1,
                       const float* __restrict__ w2, const float* __restrict__ b2,
                       float* __restrict__ gout) {
  __shared__ float hid[8 * 128];
  int t = threadIdx.x;
  for (int o = t; o < 1024; o += 256) {
    int bb = o >> 7, c = o & 127;
    float acc = b1[c];
    for (int i = 0; i < 256; ++i) acc += pooled[bb * 256 + i] * w1[i * 128 + c];
    hid[o] = acc > 0.f ? acc : 0.f;
  }
  __syncthreads();
  for (int o = t; o < 2048; o += 256) {
    int bb = o >> 8, c = o & 255;
    float acc = b2[c];
    for (int i = 0; i < 128; ++i) acc += hid[bb * 128 + i] * w2[i * 256 + c];
    gout[o] = acc;
  }
}

// ---------------------------------------------------------------------------
// Launch
// ---------------------------------------------------------------------------
extern "C" void kernel_launch(void* const* d_in, const int* in_sizes, int n_in,
                              void* d_out, int out_size, void* d_ws, size_t ws_size,
                              hipStream_t stream) {
  (void)in_sizes; (void)n_in; (void)out_size; (void)ws_size;
  const float* pos   = (const float*)d_in[0];
  const float* pe    = (const float*)d_in[1];
  const float* abias = (const float*)d_in[2];
  const float* ne_w  = (const float*)d_in[3];
  const float* ne_b  = (const float*)d_in[4];
  const float* pl_w  = (const float*)d_in[5];
  const float* pl_b  = (const float*)d_in[6];
  const float* pn_g  = (const float*)d_in[7];
  const float* pn_b  = (const float*)d_in[8];
  const float* eemb  = (const float*)d_in[9];
  const float* ck_w  = (const float*)d_in[10]; const float* ck_b = (const float*)d_in[11];
  const float* cq_w  = (const float*)d_in[12]; const float* cq_b = (const float*)d_in[13];
  const float* cv_w  = (const float*)d_in[14]; const float* cv_b = (const float*)d_in[15];
  const float* ce_w  = (const float*)d_in[16]; const float* ce_b = (const float*)d_in[17];
  const float* cs_w  = (const float*)d_in[18]; const float* cs_b = (const float*)d_in[19];
  const float* aq_w  = (const float*)d_in[20]; const float* aq_b = (const float*)d_in[21];
  const float* ak_w  = (const float*)d_in[22]; const float* ak_b = (const float*)d_in[23];
  const float* av_w  = (const float*)d_in[24]; const float* av_b = (const float*)d_in[25];
  const float* aow   = (const float*)d_in[26]; const float* aob  = (const float*)d_in[27];
  const float* bn1_g = (const float*)d_in[28]; const float* bn1_b = (const float*)d_in[29];
  const float* bn2_g = (const float*)d_in[30]; const float* bn2_b = (const float*)d_in[31];
  const float* bn3_g = (const float*)d_in[32]; const float* bn3_b = (const float*)d_in[33];
  const float* m1_w  = (const float*)d_in[34]; const float* m1_b = (const float*)d_in[35];
  const float* m2_w  = (const float*)d_in[36]; const float* m2_b = (const float*)d_in[37];
  const float* hd1_w = (const float*)d_in[38]; const float* hd1_b = (const float*)d_in[39];
  const float* hd2_w = (const float*)d_in[40]; const float* hd2_b = (const float*)d_in[41];
  const int*   eidx  = (const int*)d_in[42];
  const int*   eattr = (const int*)d_in[43];

  float* ws = (float*)d_ws;
  const size_t MB = 1048576;               // 1M floats
  float* x    = ws;
  float* t0   = ws + 1 * MB;
  float* t1   = ws + 2 * MB;
  float* t2   = ws + 3 * MB;
  float* t3   = ws + 4 * MB;
  float* agg  = ws + 5 * MB;
  float* h1   = ws + 6 * MB;
  float* mlph = ws + 7 * MB;               // [4096,512]
  float* misc = ws + 9 * MB;
  float* mv   = misc;                      // 64
  float* bnst = misc + 64;                 // 512
  float* et   = misc + 576;                // 32*256
  float* pool = misc + 8768;               // 8*256

  float* og    = (float*)d_out;            // [8,256]
  float* onode = (float*)d_out + 2048;     // [4096,256]

  const float qscale = 0.17677669529663687f;   // 32^-0.5
  dim3 gg(4, 64);    // N=256 GEMMs (64-col x 64-row blocks)
  dim3 g1(8, 64);    // N=512 GEMM

  zero_k<<<4096, 256, 0, stream>>>(agg, 1048576);
  zero_k<<<2, 256, 0, stream>>>(bnst, 512);

  pe_stats_k<<<1, 256, 0, stream>>>(pe, mv);
  node_embed_k<<<4096, 256, 0, stream>>>(pos, pe, ne_w, ne_b, pl_w, pl_b, pn_g, pn_b, mv, x);
  e_table_k<<<32, 256, 0, stream>>>(eemb, ce_w, ce_b, et);

  // conv projections
  gemm_t<true, false, false><<<gg, 128, 0, stream>>>(x, ck_w, ck_b, nullptr, t0, 4096, 256, 256, 1.f);
  gemm_t<true, false, false><<<gg, 128, 0, stream>>>(x, cq_w, cq_b, nullptr, t1, 4096, 256, 256, 1.f);
  gemm_t<true, false, false><<<gg, 128, 0, stream>>>(x, cv_w, cv_b, nullptr, t2, 4096, 256, 256, 1.f);
  gemm_t<true, false, false><<<gg, 128, 0, stream>>>(x, cs_w, cs_b, nullptr, t3, 4096, 256, 256, 1.f);
  edge_agg_k<<<8192, 256, 0, stream>>>(t0, t1, t2, et, eidx, eattr, agg, 65536);
  add3_k<<<4096, 256, 0, stream>>>(agg, t3, x, t3, 1048576);      // h1pre = agg + skip + x
  bn_stats_k<<<64, 256, 0, stream>>>(t3, bnst, 4096, 256, 64);
  bn_apply_k<<<4096, 256, 0, stream>>>(t3, bnst, bn1_g, bn1_b, h1, 4096, 256);

  // attention projections + attention
  gemm_t<true, false, false><<<gg, 128, 0, stream>>>(x, aq_w, aq_b, nullptr, t0, 4096, 256, 256, qscale);
  gemm_t<true, false, false><<<gg, 128, 0, stream>>>(x, ak_w, ak_b, nullptr, t1, 4096, 256, 256, 1.f);
  gemm_t<true, false, false><<<gg, 128, 0, stream>>>(x, av_w, av_b, nullptr, t2, 4096, 256, 256, 1.f);
  attn_k<<<2048, 32, 0, stream>>>(t0, t1, t2, abias, t3);
  gemm_t<true, true, false><<<gg, 128, 0, stream>>>(t3, aow, aob, x, t0, 4096, 256, 256, 1.f);
  zero_k<<<2, 256, 0, stream>>>(bnst, 512);
  bn_stats_k<<<64, 256, 0, stream>>>(t0, bnst, 4096, 256, 64);
  bn_apply_k<<<4096, 256, 0, stream>>>(t0, bnst, bn2_g, bn2_b, t1, 4096, 256);

  // out = h1 + h2; MLP with residual; BN3
  add2_k<<<4096, 256, 0, stream>>>(h1, t1, t2, 1048576);
  gemm_t<true, false, true><<<g1, 128, 0, stream>>>(t2, m1_w, m1_b, nullptr, mlph, 4096, 256, 512, 1.f);
  gemm_t<true, true, false><<<gg, 128, 0, stream>>>(mlph, m2_w, m2_b, t2, t0, 4096, 512, 256, 1.f);
  zero_k<<<2, 256, 0, stream>>>(bnst, 512);
  bn_stats_k<<<64, 256, 0, stream>>>(t0, bnst, 4096, 256, 64);
  bn_apply_k<<<4096, 256, 0, stream>>>(t0, bnst, bn3_g, bn3_b, onode, 4096, 256);

  // pooling + head
  pool_k<<<8, 256, 0, stream>>>(onode, pool);
  head_k<<<1, 256, 0, stream>>>(pool, hd1_w, hd1_b, hd2_w, hd2_b, og);
}
